// MPAMC_41137196761620
// MI455X (gfx1250) — matrix operand order repaired and checked by reading, not compiled, not run on hardware
//
#include <hip/hip_runtime.h>
#include <math.h>
#include <stdint.h>

// ---------------------------------------------------------------------------
// Problem constants (from setup_inputs)
// ---------------------------------------------------------------------------
constexpr int Bc   = 2048;   // batch
constexpr int Dc   = 2048;   // feature dim
constexpr int Hc   = 1024;   // hidden
constexpr int Bd   = 512;    // boundary dim
constexpr int Lc   = 256;    // latent dim
constexpr int Nc   = 65536;  // memory rows
constexpr int KTOP = 3;
constexpr int OUTW = Dc + Lc + KTOP;   // 2307
constexpr int NSLICE  = 32;
constexpr int SLICE_N = Nc / NSLICE;   // 2048 cols per slice

// sims kernel tiling: 4 waves/WG, 64 rows/WG, B tiles of 32 cols staged in LDS
constexpr int TILE_COLS  = 32;
constexpr int LDSROW     = Lc + 16;            // 256 f16 + 8 DWORD pad = 272 halves (544 B)
constexpr int STAGE_BYTES = TILE_COLS * LDSROW * 2;  // 17408
constexpr int NTILES     = SLICE_N / TILE_COLS;      // 64

typedef __attribute__((ext_vector_type(16))) _Float16 v16h;
typedef __attribute__((ext_vector_type(8)))  _Float16 v8h;
typedef __attribute__((ext_vector_type(8)))  float    v8f;
typedef __attribute__((ext_vector_type(4)))  unsigned int u32x4;
typedef __attribute__((ext_vector_type(4)))  int      i32x4;
typedef __attribute__((ext_vector_type(8)))  int      i32x8;

union AFrag { v16h v; v8h h[2]; };

// Tensor Data Mover path: device pass only (host pass sees x86 target where
// __has_builtin would be false) with plain-copy fallback.
#if defined(__HIP_DEVICE_COMPILE__) && \
    __has_builtin(__builtin_amdgcn_tensor_load_to_lds) && \
    __has_builtin(__builtin_amdgcn_s_wait_tensorcnt)
#define SIMS_USE_TDM 1
#else
#define SIMS_USE_TDM 0
#endif

#if defined(__HIP_DEVICE_COMPILE__)
// Launder a uniform value through an SGPR so the optimizer can't treat it as
// a ConstantExpr (ptrtoint of an LDS global crashes ConstantFoldBinaryOp in
// AMDGPUCodeGenPrepare on this toolchain).  "+s" keeps it scalar so TDM
// descriptor fields stay in SGPRs.
__device__ __forceinline__ unsigned opaque_sgpr(unsigned v) {
    asm volatile("" : "+s"(v));
    return v;
}
#else
__device__ __forceinline__ unsigned opaque_sgpr(unsigned v) { return v; }
#endif

__device__ __forceinline__ bool betterCand(float v, int i, float bv, int bi) {
    return (v > bv) || (v == bv && i < bi);
}

__device__ __forceinline__ void ins3(float v, int i, float* tv, int* ti) {
    if (betterCand(v, i, tv[0], ti[0])) {
        tv[2] = tv[1]; ti[2] = ti[1];
        tv[1] = tv[0]; ti[1] = ti[0];
        tv[0] = v;     ti[0] = i;
    } else if (betterCand(v, i, tv[1], ti[1])) {
        tv[2] = tv[1]; ti[2] = ti[1];
        tv[1] = v;     ti[1] = i;
    } else if (betterCand(v, i, tv[2], ti[2])) {
        tv[2] = v;     ti[2] = i;
    }
}

template<int ACT>
__device__ __forceinline__ float applyAct(float x) {
    if constexpr (ACT == 1)      return x > 0.f ? x : 0.01f * x;   // leaky relu
    else if constexpr (ACT == 2) return tanhf(x);                   // tanh
    else if constexpr (ACT == 3) return 1.f / (1.f + expf(-x));     // sigmoid
    else                         return x;
}

// ---------------------------------------------------------------------------
// Elementwise f32 -> f16 convert
// ---------------------------------------------------------------------------
__global__ void f32_to_f16_kernel(const float* __restrict__ X,
                                  _Float16* __restrict__ Y, int n) {
    int i = blockIdx.x * blockDim.x + threadIdx.x;
    if (i < n) Y[i] = (_Float16)X[i];
}

// W[K][N] f32 row-major  ->  Wt[N][K] f16 (so WMMA B fragments are contiguous)
__global__ void transpose_to_f16_kernel(const float* __restrict__ W,
                                        _Float16* __restrict__ Wt,
                                        int K, int Ncols) {
    int i = blockIdx.x * blockDim.x + threadIdx.x;
    if (i < K * Ncols) {
        int k = i / Ncols;
        int n = i - k * Ncols;
        Wt[(size_t)n * K + k] = (_Float16)W[i];
    }
}

// Row L2-normalize (f32 in, f16 out).  cols == blockDim.x (256).
__global__ void normalize_rows_kernel(const float* __restrict__ X,
                                      _Float16* __restrict__ Y, int cols) {
    __shared__ float red[256];
    int row = blockIdx.x, t = threadIdx.x;
    float x = X[(size_t)row * cols + t];
    red[t] = x * x;
    __syncthreads();
    for (int s = 128; s > 0; s >>= 1) {
        if (t < s) red[t] += red[t + s];
        __syncthreads();
    }
    float inv = 1.f / fmaxf(sqrtf(red[0]), 1e-10f);
    Y[(size_t)row * cols + t] = (_Float16)(x * inv);
}

// ---------------------------------------------------------------------------
// Fused GEMM:  C = act(A[M][K] * Bt[N][K]^T + bias[N])
// One wave per 16(M) x 64(N) tile; f16 operands, f32 WMMA accumulation.
// ---------------------------------------------------------------------------
template<int ACT, bool OUTF16>
__global__ __launch_bounds__(32) void gemm_wmma_kernel(
        const _Float16* __restrict__ A, const _Float16* __restrict__ Bt,
        const float* __restrict__ bias, void* __restrict__ Cout,
        int K, int ldC, int colOff) {
    int lane  = threadIdx.x;
    int mbase = blockIdx.y * 16;
    int nbase = blockIdx.x * 64;
    int l15   = lane & 15;
    int hi    = lane >> 4;
    int row   = mbase + l15;
    int koff  = hi * 16;
    int kb    = hi * 16;

    v8f acc[4] = {};
    for (int k0 = 0; k0 < K; k0 += 32) {
        AFrag a;
        a.h[0] = *(const v8h*)(A + (size_t)row * K + k0 + koff);
        a.h[1] = *(const v8h*)(A + (size_t)row * K + k0 + 8 + koff);
#pragma unroll
        for (int t = 0; t < 4; ++t) {
            int col = nbase + t * 16 + l15;
            v16h b = *(const v16h*)(Bt + (size_t)col * K + k0 + kb);
            acc[t] = __builtin_amdgcn_wmma_f32_16x16x32_f16(
                false, a.v, false, b, (short)0, acc[t], false, false);
        }
    }

    int rlo = hi * 8;
#pragma unroll
    for (int t = 0; t < 4; ++t) {
        int col = nbase + t * 16 + l15;
        float bz = bias[col];
#pragma unroll
        for (int v = 0; v < 8; ++v) {
            float xv = applyAct<ACT>(acc[t][v] + bz);
            int orow = mbase + rlo + v;
            if constexpr (OUTF16)
                ((_Float16*)Cout)[(size_t)orow * ldC + col] = (_Float16)xv;
            else
                ((float*)Cout)[(size_t)orow * ldC + colOff + col] = xv;
        }
    }
}

// ---------------------------------------------------------------------------
// TDM: issue a 2D tensor load of TILE_COLS rows x 256 f16 from memh into LDS,
// padding each 512B row with 8 DWORDs (LDS row stride 544 B = LDSROW halves).
// ---------------------------------------------------------------------------
#if SIMS_USE_TDM
__device__ __forceinline__ void tdm_issue_tile(const _Float16* gsrc, unsigned lds_off) {
    union { unsigned long long q[2]; u32x4 v; } g0 = {};
    union { unsigned long long q[4]; i32x8 v; } g1 = {};
    i32x4 g2 = {0, 0, 0, 0};
    i32x4 g3 = {0, 0, 0, 0};
    unsigned long long ga = (unsigned long long)(uintptr_t)gsrc;
    // group0: count=1 | lds_addr[63:32] ; global_addr[120:64] | type=2 [127:126]
    g0.q[0] = 1ull | ((unsigned long long)lds_off << 32);
    g0.q[1] = (ga & 0x01FFFFFFFFFFFFFFull) | (2ull << 62);
    // group1: data_size=2B [17:16]; pad_enable [20]; pad_interval=128DW (code 6)
    //         [24:22]; pad_amount=8DW (code 7) [31:25]; tensor_dim0=256 [79:48]
    g1.q[0] = (1ull << 16) | (1ull << 20) | (6ull << 22) | (7ull << 25)
            | (((unsigned long long)Lc & 0xFFFFull) << 48);
    // tensor_dim1=Nc [111:80]; tile_dim0=256 [127:112]
    g1.q[1] = ((unsigned long long)Lc >> 16)
            | ((unsigned long long)Nc << 16)
            | ((unsigned long long)Lc << 48);
    // tile_dim1=TILE_COLS [143:128]; tensor_dim0_stride=256 [207:160]
    g1.q[2] = (unsigned long long)TILE_COLS
            | ((unsigned long long)Lc << 32);
    g1.q[3] = 0;   // tensor_dim1_stride unused (2D tile)
#if __clang_major__ >= 23
    i32x8 g4 = {0, 0, 0, 0, 0, 0, 0, 0};
    __builtin_amdgcn_tensor_load_to_lds(g0.v, g1.v, g2, g3, g4, 0);
#else
    __builtin_amdgcn_tensor_load_to_lds(g0.v, g1.v, g2, g3, 0);
#endif
}
#endif

// Fallback staging: cooperative b128 copy honoring the padded LDS layout.
__device__ __forceinline__ void stage_copy_tile(_Float16* buf, const _Float16* gsrc) {
    int tid = threadIdx.x;
    for (int idx = tid; idx < TILE_COLS * 16; idx += 128) {
        int r = idx >> 4, s = idx & 15;
        *(v16h*)(buf + (size_t)r * LDSROW + s * 16) =
            *(const v16h*)(gsrc + (size_t)r * Lc + s * 16);
    }
}

// ---------------------------------------------------------------------------
// Fused cosine-sims GEMM + running top-3.
// Grid (NSLICE, Bc/64); 4 waves per WG, 16 rows per wave.  A (qn rows) is
// preloaded to registers once; B tiles (32 cols x 256 k of memh) are staged
// into LDS via the Tensor Data Mover (double-buffered) and shared by all
// four waves, cutting L2 traffic 4x vs per-wave streaming.
// ---------------------------------------------------------------------------
__global__ __launch_bounds__(128) void sims_topk_kernel(
        const _Float16* __restrict__ qn, const _Float16* __restrict__ memh,
        float* __restrict__ pv, int* __restrict__ pi) {
    __shared__ __align__(32) unsigned char smem[2 * STAGE_BYTES];

    const int tid   = threadIdx.x;
    const int lane  = tid & 31;
    const int wid   = tid >> 5;
    const int l15   = lane & 15;
    const int hi    = lane >> 4;
    const int slice = blockIdx.x;
    const int rbase = blockIdx.y * 64 + wid * 16;   // wave's 16-row strip
    const int row   = rbase + l15;
    const int koff  = hi * 16;
    const int kb    = hi * 16;

    // Preload this wave's entire A operand (16 rows x 256 k) into registers.
    AFrag a[8];
#pragma unroll
    for (int kc = 0; kc < 8; ++kc) {
        a[kc].h[0] = *(const v8h*)(qn + (size_t)row * Lc + kc * 32 + koff);
        a[kc].h[1] = *(const v8h*)(qn + (size_t)row * Lc + kc * 32 + 8 + koff);
    }

    float tv[8][3];
    int   ti[8][3];
#pragma unroll
    for (int v = 0; v < 8; ++v)
#pragma unroll
        for (int j = 0; j < 3; ++j) { tv[v][j] = -3.4e38f; ti[v][j] = 0x7fffffff; }

    const int n_begin = slice * SLICE_N;
#if SIMS_USE_TDM
    // Laundered through an SGPR: must not be a ConstantExpr (see opaque_sgpr).
    const unsigned lds_base = opaque_sgpr((unsigned)(uintptr_t)smem);
    if (tid < 32) {
        tdm_issue_tile(memh + (size_t)n_begin * Lc, lds_base);
        __builtin_amdgcn_s_wait_tensorcnt(0);
    }
#else
    stage_copy_tile((_Float16*)smem, memh + (size_t)n_begin * Lc);
#endif
    __syncthreads();

    for (int tile = 0; tile < NTILES; ++tile) {
        const int n0 = n_begin + tile * TILE_COLS;
        // Prefetch next tile into the other buffer while computing this one.
        if (tile + 1 < NTILES) {
#if SIMS_USE_TDM
            if (tid < 32)
                tdm_issue_tile(memh + (size_t)(n0 + TILE_COLS) * Lc,
                               lds_base + (unsigned)(((tile + 1) & 1) * STAGE_BYTES));
#else
            stage_copy_tile((_Float16*)(smem + ((tile + 1) & 1) * STAGE_BYTES),
                            memh + (size_t)(n0 + TILE_COLS) * Lc);
#endif
        }

        // No pointer array: select an integer offset, then index (plain GEP).
        const _Float16* ldsB = (const _Float16*)(smem + (tile & 1) * STAGE_BYTES);
        v8f acc[2] = {};
#pragma unroll
        for (int kc = 0; kc < 8; ++kc) {
#pragma unroll
            for (int t = 0; t < 2; ++t) {
                v16h b = *(const v16h*)(ldsB + (size_t)(t * 16 + l15) * LDSROW + kc * 32 + kb);
                acc[t] = __builtin_amdgcn_wmma_f32_16x16x32_f16(
                    false, a[kc].v, false, b, (short)0, acc[t], false, false);
            }
        }
#pragma unroll
        for (int t = 0; t < 2; ++t) {
            int col = n0 + t * 16 + l15;
#pragma unroll
            for (int v = 0; v < 8; ++v) ins3(acc[t][v], col, tv[v], ti[v]);
        }

#if SIMS_USE_TDM
        if (tile + 1 < NTILES && tid < 32) __builtin_amdgcn_s_wait_tensorcnt(0);
#endif
        __syncthreads();
    }

    // Merge per-lane top-3 candidates per row.  Scratch overlays the staging
    // buffers (all compute is done); each wave uses a private 16x48 region.
    float* mlv = (float*)smem + (size_t)wid * 16 * 48;
    int*   mli = (int*)(smem + 4 * 16 * 48 * sizeof(float)) + (size_t)wid * 16 * 48;
#pragma unroll
    for (int v = 0; v < 8; ++v) {
        int r = v + hi * 8;
#pragma unroll
        for (int j = 0; j < 3; ++j) {
            mlv[r * 48 + l15 * 3 + j] = tv[v][j];
            mli[r * 48 + l15 * 3 + j] = ti[v][j];
        }
    }
    __syncthreads();
    if (lane < 16) {
        float bv[3] = {-3.4e38f, -3.4e38f, -3.4e38f};
        int   bi[3] = {0x7fffffff, 0x7fffffff, 0x7fffffff};
        for (int s = 0; s < 48; ++s) ins3(mlv[lane * 48 + s], mli[lane * 48 + s], bv, bi);
        int r = rbase + lane;
        for (int j = 0; j < 3; ++j) {
            pv[((size_t)r * NSLICE + slice) * 3 + j] = bv[j];
            pi[((size_t)r * NSLICE + slice) * 3 + j] = bi[j];
        }
    }
}

// ---------------------------------------------------------------------------
// Finalize: merge slice partials -> top-3, softmax readout, exp_map, output.
// One 64-thread block per batch row; 4 latent elems per thread.
// ---------------------------------------------------------------------------
__device__ float blockReduceSum64(float x, float* red) {
    int t = threadIdx.x;
    __syncthreads();
    red[t] = x;
    __syncthreads();
    for (int s = 32; s > 0; s >>= 1) {
        if (t < s) red[t] += red[t + s];
        __syncthreads();
    }
    return red[0];
}

__global__ __launch_bounds__(64) void finalize_kernel(
        const float* __restrict__ memory, const float* __restrict__ pv,
        const int* __restrict__ pi, const float* __restrict__ hyp_base,
        float* __restrict__ out) {
    int row = blockIdx.x, t = threadIdx.x;
    __shared__ float sv[3];
    __shared__ int   si[3];
    __shared__ float red[64];

    if (t == 0) {
        float bv[3] = {-3.4e38f, -3.4e38f, -3.4e38f};
        int   bi[3] = {0x7fffffff, 0x7fffffff, 0x7fffffff};
        const float* pvr = pv + (size_t)row * NSLICE * 3;
        const int*   pir = pi + (size_t)row * NSLICE * 3;
        for (int s = 0; s < NSLICE * 3; ++s) ins3(pvr[s], pir[s], bv, bi);
        for (int j = 0; j < 3; ++j) { sv[j] = bv[j]; si[j] = bi[j]; }
    }
    __syncthreads();

    float m  = sv[0];                       // sorted desc -> max first
    float e0 = expf(sv[0] - m), e1 = expf(sv[1] - m), e2 = expf(sv[2] - m);
    float es = e0 + e1 + e2;
    float w0 = e0 / es, w1 = e1 / es, w2 = e2 / es;

    const float* m0 = memory + (size_t)si[0] * Lc;
    const float* m1 = memory + (size_t)si[1] * Lc;
    const float* m2 = memory + (size_t)si[2] * Lc;

    float vv[4], xb[4], yy[4];
    float vsq_l = 0.f;
#pragma unroll
    for (int q = 0; q < 4; ++q) {
        int e = t * 4 + q;
        float r = w0 * m0[e] + w1 * m1[e] + w2 * m2[e];   // soft readout
        float v = 0.1f * r;                               // tangent vector
        vv[q] = v;
        vsq_l += v * v;
        xb[q] = hyp_base[e];
    }
    float vsq  = blockReduceSum64(vsq_l, red);
    float vn   = fmaxf(sqrtf(vsq), 1e-10f);
    float coef = tanhf(0.5f * vn) / vn;                   // sqrt(-c)=1

    float xy_l = 0.f, xs_l = 0.f, ys_l = 0.f;
#pragma unroll
    for (int q = 0; q < 4; ++q) {
        yy[q] = vv[q] * coef;
        xy_l += xb[q] * yy[q];
        xs_l += xb[q] * xb[q];
        ys_l += yy[q] * yy[q];
    }
    float xy = blockReduceSum64(xy_l, red);
    float xs = blockReduceSum64(xs_l, red);
    float ys = blockReduceSum64(ys_l, red);

    // Mobius add with c = -1
    float den = 1.f - 2.f * xy + xs * ys;
    float* orow = out + (size_t)row * OUTW;
#pragma unroll
    for (int q = 0; q < 4; ++q) {
        int e = t * 4 + q;
        float num = (1.f - 2.f * xy - ys) * xb[q] + (1.f + xs) * yy[q];
        orow[Dc + e] = num / den;
    }
    if (t < 3) orow[Dc + Lc + t] = sv[t];
}

// ---------------------------------------------------------------------------
// Host launcher
// ---------------------------------------------------------------------------
extern "C" void kernel_launch(void* const* d_in, const int* in_sizes, int n_in,
                              void* d_out, int out_size, void* d_ws, size_t ws_size,
                              hipStream_t stream) {
    (void)in_sizes; (void)n_in; (void)out_size; (void)ws_size;

    const float* x        = (const float*)d_in[0];
    const float* memory   = (const float*)d_in[1];
    const float* enc_w1   = (const float*)d_in[2];
    const float* enc_b1   = (const float*)d_in[3];
    const float* enc_w2   = (const float*)d_in[4];
    const float* enc_b2   = (const float*)d_in[5];
    const float* lat_w    = (const float*)d_in[6];
    const float* lat_b    = (const float*)d_in[7];
    const float* dec_w1   = (const float*)d_in[8];
    const float* dec_b1   = (const float*)d_in[9];
    const float* dec_w2   = (const float*)d_in[10];
    const float* dec_b2   = (const float*)d_in[11];
    const float* hyp_base = (const float*)d_in[12];
    float* out = (float*)d_out;

    char* wptr = (char*)d_ws;
    auto alloc = [&](size_t bytes) -> void* {
        void* p = (void*)wptr;
        wptr += (bytes + 255) & ~(size_t)255;
        return p;
    };
    _Float16* xh     = (_Float16*)alloc((size_t)Bc * Dc * 2);
    _Float16* W1t    = (_Float16*)alloc((size_t)Dc * Hc * 2);
    _Float16* W2t    = (_Float16*)alloc((size_t)Hc * Bd * 2);
    _Float16* LWt    = (_Float16*)alloc((size_t)Bd * Lc * 2);
    _Float16* D1t    = (_Float16*)alloc((size_t)Bd * Hc * 2);
    _Float16* D2t    = (_Float16*)alloc((size_t)Hc * Dc * 2);
    _Float16* h16    = (_Float16*)alloc((size_t)Bc * Hc * 2);
    _Float16* bd16   = (_Float16*)alloc((size_t)Bc * Bd * 2);
    _Float16* t16    = (_Float16*)alloc((size_t)Bc * Hc * 2);
    float*    latent = (float*)   alloc((size_t)Bc * Lc * 4);
    _Float16* qnh    = (_Float16*)alloc((size_t)Bc * Lc * 2);
    _Float16* memh   = (_Float16*)alloc((size_t)Nc * Lc * 2);
    float*    pv     = (float*)   alloc((size_t)Bc * NSLICE * 3 * 4);
    int*      pi     = (int*)     alloc((size_t)Bc * NSLICE * 3 * 4);

    // 1) operand conversion / weight transposition to f16
    {
        int n = Bc * Dc;
        f32_to_f16_kernel<<<(n + 255) / 256, 256, 0, stream>>>(x, xh, n);
    }
    transpose_to_f16_kernel<<<(Dc * Hc + 255) / 256, 256, 0, stream>>>(enc_w1, W1t, Dc, Hc);
    transpose_to_f16_kernel<<<(Hc * Bd + 255) / 256, 256, 0, stream>>>(enc_w2, W2t, Hc, Bd);
    transpose_to_f16_kernel<<<(Bd * Lc + 255) / 256, 256, 0, stream>>>(lat_w, LWt, Bd, Lc);
    transpose_to_f16_kernel<<<(Bd * Hc + 255) / 256, 256, 0, stream>>>(dec_w1, D1t, Bd, Hc);
    transpose_to_f16_kernel<<<(Hc * Dc + 255) / 256, 256, 0, stream>>>(dec_w2, D2t, Hc, Dc);

    // 2) fused WMMA GEMM chain
    dim3 blk(32, 1, 1);
    gemm_wmma_kernel<1, true ><<<dim3(Hc / 64, Bc / 16), blk, 0, stream>>>(
        xh, W1t, enc_b1, h16, Dc, Hc, 0);                 // h = leaky(x@W1+b1)
    gemm_wmma_kernel<2, true ><<<dim3(Bd / 64, Bc / 16), blk, 0, stream>>>(
        h16, W2t, enc_b2, bd16, Hc, Bd, 0);               // boundary = tanh(h@W2+b2)
    gemm_wmma_kernel<0, false><<<dim3(Lc / 64, Bc / 16), blk, 0, stream>>>(
        bd16, LWt, lat_b, latent, Bd, Lc, 0);             // latent = boundary@LW+b
    gemm_wmma_kernel<1, true ><<<dim3(Hc / 64, Bc / 16), blk, 0, stream>>>(
        bd16, D1t, dec_b1, t16, Bd, Hc, 0);               // t = leaky(boundary@D1+b)
    gemm_wmma_kernel<3, false><<<dim3(Dc / 64, Bc / 16), blk, 0, stream>>>(
        t16, D2t, dec_b2, out, Hc, OUTW, 0);              // recon -> out[:, 0:2048]

    // 3) cosine normalization (folded into f16 operands)
    normalize_rows_kernel<<<Bc, 256, 0, stream>>>(latent, qnh, Lc);
    normalize_rows_kernel<<<Nc, 256, 0, stream>>>(memory, memh, Lc);

    // 4) sims GEMM (TDM-staged B, register-resident A) + per-slice top-3
    sims_topk_kernel<<<dim3(NSLICE, Bc / 64), dim3(128, 1, 1), 0, stream>>>(
        qnh, memh, pv, pi);

    // 5) merge top-3, softmax readout, exp_map -> out[:, 2048:2307]
    finalize_kernel<<<Bc, 64, 0, stream>>>(memory, pv, pi, hyp_base, out);
}
